// DualTopic_54846732370327
// MI455X (gfx1250) — matrix operand order repaired and checked by reading, not compiled, not run on hardware
//
#include <hip/hip_runtime.h>
#include <cmath>

// ---------------------------------------------------------------------------
// Problem constants (match reference)
// ---------------------------------------------------------------------------
#define BSZ   512
#define V_EN  30000
#define V_CN  20000
#define KTOP  100
#define HDIM  512
#define DEMB  300
#define TAU_F       1.0f
#define EPS_SINK    0.1f
#define SINK_ITERS  50

// ---------------------------------------------------------------------------
// bf16 helpers
// ---------------------------------------------------------------------------
typedef __bf16 bf16_t;
typedef bf16_t v16bf __attribute__((ext_vector_type(16)));
typedef float  v8f   __attribute__((ext_vector_type(8)));
typedef unsigned short us4 __attribute__((ext_vector_type(4)));

__device__ __forceinline__ unsigned short f32_to_bf16_bits(float f) {
  unsigned int u = __builtin_bit_cast(unsigned int, f);
  u = u + 0x7FFFu + ((u >> 16) & 1u);   // round-to-nearest-even
  return (unsigned short)(u >> 16);
}

__device__ __forceinline__ float softplus_f(float x) {
  return (x > 0.f) ? (x + log1pf(__expf(-x))) : log1pf(__expf(x));
}

// ---------------------------------------------------------------------------
// Generic WMMA GEMM: Out = epilogue(A[MxK] @ B[KxN] + bias)
//   block = 256 threads = 8 waves; wave computes one 16x16 tile,
//   block tile = 32 (M) x 64 (N), K staged in LDS in steps of 32 (bf16).
//
// LDS tiles are stored PRE-SWIZZLED into the WMMA fragment layout
// (ISA 7.12.2) so the per-wave fragment load is 2x ds_load_b128 per operand:
//   A (16-bit 16x32):  element (m,k) -> lane = m + 16*((k>>3)&1),
//                                       e    = (k&7) | ((k>>4)<<3)
//   B (16-bit 32x16):  element (k,n) -> lane = n + 16*((k>>3)&1),
//                                       e    = (k&7) | ((k>>4)<<3)
//
// Interior blocks run a DOUBLE-BUFFERED pipeline: one barrier per K-tile,
// next tile's global loads overlap the current v_wmma. Edge blocks and the
// K tail take a guarded single-buffer path.
//
// epilogue modes (compile-time):
//   0: plain (+bias if non-null)
//   1: softplus(acc + bias)
//   2: beta:  sq = aux_m[m] + aux_n[n] - 2*acc; e = exp(-sqrt(max(sq,0))*scale)
//             stored TRANSPOSED: Out[n*ldo + m]
//   3: exp(acc * scale)
// ---------------------------------------------------------------------------
#define BM 32
#define BN 64
#define BK 32

template<int GUARD, int TRANSB>
__device__ __forceinline__ void stage_AB(const float* __restrict__ A, int lda,
                                         const float* __restrict__ Bm, int ldb,
                                         int M, int N, int Kd,
                                         int bm0, int bn0, int k0,
                                         int vecA, int vecB,
                                         unsigned short* __restrict__ asu,
                                         unsigned short* __restrict__ bsu,
                                         int tid)
{
  // ---- A tile: thread handles row r = tid>>3, 4 consecutive k
  {
    const int r   = tid >> 3;          // 0..31
    const int kk0 = (tid & 7) << 2;    // 0,4,...,28
    const int gm  = bm0 + r;
    const int m   = r & 15;
    const int wbase = ((r >> 4) * 32) << 4;            // waveM*32*16
    const int la = m + (((kk0 >> 3) & 1) << 4);        // same for all 4 k
    const int e0 = (kk0 & 7) | ((kk0 >> 4) << 3);      // consecutive e
    us4 pk;
    if (GUARD) {
#pragma unroll
      for (int q = 0; q < 4; ++q) {
        int gk = k0 + kk0 + q;
        float v = (gm < M && gk < Kd) ? A[(long)gm * lda + gk] : 0.f;
        pk[q] = f32_to_bf16_bits(v);
      }
    } else if (vecA) {
      float4 av = *(const float4*)(A + (long)gm * lda + (k0 + kk0));
      pk[0] = f32_to_bf16_bits(av.x);
      pk[1] = f32_to_bf16_bits(av.y);
      pk[2] = f32_to_bf16_bits(av.z);
      pk[3] = f32_to_bf16_bits(av.w);
    } else {
      const float* ap = A + (long)gm * lda + (k0 + kk0);
#pragma unroll
      for (int q = 0; q < 4; ++q) pk[q] = f32_to_bf16_bits(ap[q]);
    }
    *(us4*)&asu[wbase + (la << 4) + e0] = pk;          // one 8B LDS store
  }
  // ---- B tile: thread handles k row kk = tid>>3, 8 consecutive n
  {
    const int kk = tid >> 3;           // 0..31
    const int c0 = (tid & 7) << 3;     // 0,8,...,56
    const int gk = k0 + kk;
    const int la_hi = ((kk >> 3) & 1) << 4;
    const int e  = (kk & 7) | ((kk >> 4) << 3);
    float vals[8];
    if (GUARD) {
#pragma unroll
      for (int q = 0; q < 8; ++q) {
        int gn = bn0 + c0 + q;
        float v = 0.f;
        if (gk < Kd && gn < N)
          v = TRANSB ? Bm[(long)gn * ldb + gk] : Bm[(long)gk * ldb + gn];
        vals[q] = v;
      }
    } else if (!TRANSB && vecB) {
      float4 b0 = *(const float4*)(Bm + (long)gk * ldb + (bn0 + c0));
      float4 b1 = *(const float4*)(Bm + (long)gk * ldb + (bn0 + c0) + 4);
      vals[0] = b0.x; vals[1] = b0.y; vals[2] = b0.z; vals[3] = b0.w;
      vals[4] = b1.x; vals[5] = b1.y; vals[6] = b1.z; vals[7] = b1.w;
    } else {
#pragma unroll
      for (int q = 0; q < 8; ++q) {
        int gn = bn0 + c0 + q;
        vals[q] = TRANSB ? Bm[(long)gn * ldb + gk] : Bm[(long)gk * ldb + gn];
      }
    }
#pragma unroll
    for (int q = 0; q < 8; ++q) {
      int c = c0 + q;
      bsu[((((c >> 4) * 32) + (c & 15) + la_hi) << 4) + e] = f32_to_bf16_bits(vals[q]);
    }
  }
}

template<int MODE, int TRANSB>
__global__ __launch_bounds__(256)
void wmma_gemm_kernel(const float* __restrict__ A, int lda,
                      const float* __restrict__ Bm, int ldb,
                      const float* __restrict__ bias,
                      float* __restrict__ Out, int ldo,
                      int M, int N, int Kd, int vecA, int vecB,
                      const float* __restrict__ aux_m,
                      const float* __restrict__ aux_n,
                      float scale)
{
  __shared__ v16bf AsF[2][2 * 32];    // double-buffered fragment-layout tiles
  __shared__ v16bf BsF[2][4 * 32];    // 12 KB total

  const int tid   = threadIdx.x;
  const int lane  = tid & 31;
  const int wave  = tid >> 5;      // 0..7
  const int waveM = wave >> 2;     // 0..1
  const int waveN = wave & 3;      // 0..3
  const int bm0   = blockIdx.y * BM;
  const int bn0   = blockIdx.x * BN;
  const int hi    = lane >> 4;
  const int l15   = lane & 15;
  const int aidx  = waveM * 32 + lane;
  const int bidx  = waveN * 32 + lane;

  v8f acc = {0.f, 0.f, 0.f, 0.f, 0.f, 0.f, 0.f, 0.f};

  const bool interior = ((bm0 + BM) <= M) && ((bn0 + BN) <= N);
  const int kfull  = Kd / BK;
  const int ktiles = (Kd + BK - 1) / BK;

  if (interior) {
    int cur = 0;
    if (kfull > 0) {
      // prologue: stage tile 0 into buffer 0
      stage_AB<0, TRANSB>(A, lda, Bm, ldb, M, N, Kd, bm0, bn0, 0,
                          vecA, vecB, (unsigned short*)AsF[0],
                          (unsigned short*)BsF[0], tid);
      __syncthreads();
      for (int kt = 0; kt < kfull; ++kt) {
        v16bf afrag = AsF[cur][aidx];
        v16bf bfrag = BsF[cur][bidx];
        if (kt + 1 < kfull) {
          // overlap next tile's global loads with this tile's wmma
          stage_AB<0, TRANSB>(A, lda, Bm, ldb, M, N, Kd, bm0, bn0,
                              (kt + 1) * BK, vecA, vecB,
                              (unsigned short*)AsF[cur ^ 1],
                              (unsigned short*)BsF[cur ^ 1], tid);
        }
        acc = __builtin_amdgcn_wmma_f32_16x16x32_bf16(
            false, afrag, false, bfrag, (short)0, acc, false, false);
        __syncthreads();
        cur ^= 1;
      }
    }
    if (kfull != ktiles) {
      // guarded K tail
      stage_AB<1, TRANSB>(A, lda, Bm, ldb, M, N, Kd, bm0, bn0, kfull * BK,
                          vecA, vecB, (unsigned short*)AsF[cur],
                          (unsigned short*)BsF[cur], tid);
      __syncthreads();
      v16bf afrag = AsF[cur][aidx];
      v16bf bfrag = BsF[cur][bidx];
      acc = __builtin_amdgcn_wmma_f32_16x16x32_bf16(
          false, afrag, false, bfrag, (short)0, acc, false, false);
    }
  } else {
    for (int kt = 0; kt < ktiles; ++kt) {
      stage_AB<1, TRANSB>(A, lda, Bm, ldb, M, N, Kd, bm0, bn0, kt * BK,
                          vecA, vecB, (unsigned short*)AsF[0],
                          (unsigned short*)BsF[0], tid);
      __syncthreads();
      v16bf afrag = AsF[0][aidx];
      v16bf bfrag = BsF[0][bidx];
      acc = __builtin_amdgcn_wmma_f32_16x16x32_bf16(
          false, afrag, false, bfrag, (short)0, acc, false, false);
      __syncthreads();
    }
  }

  // Epilogue; D layout: VGPR r -> M = r + 8*hi, N = lane&15
  const int gn = bn0 + waveN * 16 + l15;
#pragma unroll
  for (int r = 0; r < 8; ++r) {
    int gm = bm0 + waveM * 16 + r + (hi << 3);
    if (gm < M && gn < N) {
      float v = acc[r];
      if (bias) v += bias[gn];
      if (MODE == 1) {
        Out[(long)gm * ldo + gn] = softplus_f(v);
      } else if (MODE == 2) {
        float sq = aux_m[gm] + aux_n[gn] - 2.f * v;
        float d  = sqrtf(fmaxf(sq, 0.f));
        Out[(long)gn * ldo + gm] = __expf(-d * scale);
      } else if (MODE == 3) {
        Out[(long)gm * ldo + gn] = __expf(v * scale);
      } else {
        Out[(long)gm * ldo + gn] = v;
      }
    }
  }
}

// ---------------------------------------------------------------------------
// Row squared-norms: out[r] = sum_c X[r,c]^2
// ---------------------------------------------------------------------------
__global__ void rowsq_kernel(const float* __restrict__ X, int rows, int cols,
                             float* __restrict__ out)
{
  int r = blockIdx.x * blockDim.x + threadIdx.x;
  if (r < rows) {
    const float* p = X + (long)r * cols;
    float s = 0.f;
    for (int c = 0; c < cols; ++c) { float x = p[c]; s += x * x; }
    out[r] = s;
  }
}

// ---------------------------------------------------------------------------
// Reparameterize + softmax + per-row KL.  One block per batch row, 128 thr.
// Writes theta to d_out AND to a 16B-aligned workspace copy (for GEMM A use).
// ---------------------------------------------------------------------------
__global__ __launch_bounds__(128)
void reparam_kernel(const float* __restrict__ mu, const float* __restrict__ lv,
                    const float* __restrict__ eps,
                    float* __restrict__ theta_out, float* __restrict__ theta_ws,
                    float* __restrict__ klrow)
{
  __shared__ float red[128];
  const int b = blockIdx.x, t = threadIdx.x;
  const bool act = t < KTOP;
  float muv = 0.f, lvv = 0.f, z = 0.f;
  if (act) {
    muv = mu[b * KTOP + t];
    lvv = lv[b * KTOP + t];
    z = muv + eps[b * KTOP + t] * __expf(0.5f * lvv);
  }
  red[t] = act ? z : -3.0e38f;
  __syncthreads();
  for (int s = 64; s > 0; s >>= 1) { if (t < s) red[t] = fmaxf(red[t], red[t + s]); __syncthreads(); }
  float zmax = red[0]; __syncthreads();
  float ez = act ? __expf(z - zmax) : 0.f;
  red[t] = ez; __syncthreads();
  for (int s = 64; s > 0; s >>= 1) { if (t < s) red[t] += red[t + s]; __syncthreads(); }
  float denom = red[0]; __syncthreads();
  if (act) {
    float th = ez / denom;
    theta_out[b * KTOP + t] = th;
    theta_ws[b * KTOP + t]  = th;
  }
  red[t] = act ? 0.5f * (__expf(lvv) + muv * muv - 1.f - lvv) : 0.f;
  __syncthreads();
  for (int s = 64; s > 0; s >>= 1) { if (t < s) red[t] += red[t + s]; __syncthreads(); }
  if (t == 0) klrow[b] = red[0];
}

// ---------------------------------------------------------------------------
// Normalize each beta row (topic) by its vocab sum, in place. Block per topic.
// ---------------------------------------------------------------------------
__global__ __launch_bounds__(256)
void beta_norm_kernel(float* __restrict__ beta, int V)
{
  __shared__ float red[256];
  const int t = threadIdx.x;
  float* row = beta + (long)blockIdx.x * V;
  float s = 0.f;
  for (int v = t; v < V; v += 256) s += row[v];
  red[t] = s; __syncthreads();
  for (int st = 128; st > 0; st >>= 1) { if (t < st) red[t] += red[t + st]; __syncthreads(); }
  float inv = 1.f / red[0];
  for (int v = t; v < V; v += 256) row[v] *= inv;
}

// ---------------------------------------------------------------------------
// Per-row recon loss: -(sum x*recon - (max + log(sum exp(recon-max))) * sum x)
// One block per batch row.
// ---------------------------------------------------------------------------
__global__ __launch_bounds__(256)
void recon_loss_kernel(const float* __restrict__ recon,
                       const float* __restrict__ xbow,
                       int V, float* __restrict__ lossrow)
{
  __shared__ float red[256];
  const int b = blockIdx.x, t = threadIdx.x;
  const float* r  = recon + (long)b * V;
  const float* xb = xbow  + (long)b * V;
  float mx = -3.0e38f;
  for (int v = t; v < V; v += 256) mx = fmaxf(mx, r[v]);
  red[t] = mx; __syncthreads();
  for (int s = 128; s > 0; s >>= 1) { if (t < s) red[t] = fmaxf(red[t], red[t + s]); __syncthreads(); }
  float rowmax = red[0]; __syncthreads();
  float se = 0.f, dot = 0.f, sx = 0.f;
  for (int v = t; v < V; v += 256) {
    float rv = r[v], xv = xb[v];
    se += __expf(rv - rowmax); dot += xv * rv; sx += xv;
  }
  red[t] = se; __syncthreads();
  for (int s = 128; s > 0; s >>= 1) { if (t < s) red[t] += red[t + s]; __syncthreads(); }
  float se_t = red[0]; __syncthreads();
  red[t] = dot; __syncthreads();
  for (int s = 128; s > 0; s >>= 1) { if (t < s) red[t] += red[t + s]; __syncthreads(); }
  float dot_t = red[0]; __syncthreads();
  red[t] = sx; __syncthreads();
  for (int s = 128; s > 0; s >>= 1) { if (t < s) red[t] += red[t + s]; __syncthreads(); }
  if (t == 0) {
    float lse = rowmax + logf(se_t);
    lossrow[b] = -(dot_t - lse * red[0]);
  }
}

// ---------------------------------------------------------------------------
// Contrastive: select anchor theta by parity, L2-normalize rows.
// ---------------------------------------------------------------------------
__global__ __launch_bounds__(128)
void contrast_select_kernel(const float* __restrict__ theta_en,
                            const float* __restrict__ theta_cn,
                            float* __restrict__ tn)
{
  __shared__ float red[128];
  const int b = blockIdx.x, t = threadIdx.x;
  const float* src = ((b & 1) == 0) ? theta_en : theta_cn;
  float v = (t < KTOP) ? src[b * KTOP + t] : 0.f;
  red[t] = v * v; __syncthreads();
  for (int s = 64; s > 0; s >>= 1) { if (t < s) red[t] += red[t + s]; __syncthreads(); }
  float nrm = fmaxf(sqrtf(red[0]), 1e-8f);
  if (t < KTOP) tn[b * KTOP + t] = v / nrm;
}

// ---------------------------------------------------------------------------
// Contrastive per-row loss from E = exp(sim/tau). One block per row i.
// ---------------------------------------------------------------------------
__global__ __launch_bounds__(256)
void contrast_loss_kernel(const float* __restrict__ E,
                          const int* __restrict__ cid,
                          float* __restrict__ per_row,
                          float* __restrict__ has_row)
{
  __shared__ float rp[256], rn[256], rc[256];
  const int i = blockIdx.x, t = threadIdx.x;
  const int ci = cid[i];
  const bool vi = ci > 0;
  float pos = 0.f, neg = 0.f, pcnt = 0.f;
  for (int j = t; j < BSZ; j += 256) {
    if (j == i) continue;
    float e = E[(long)i * BSZ + j];
    neg += e;
    int cj = cid[j];
    if (vi && cj > 0 && cj == ci) { pos += e; pcnt += 1.f; }
  }
  rp[t] = pos; rn[t] = neg; rc[t] = pcnt; __syncthreads();
  for (int s = 128; s > 0; s >>= 1) {
    if (t < s) { rp[t] += rp[t + s]; rn[t] += rn[t + s]; rc[t] += rc[t + s]; }
    __syncthreads();
  }
  if (t == 0) {
    bool has = vi && (rc[0] > 0.f);
    per_row[i] = has ? -logf(rp[0] / (rp[0] + rn[0] + 1e-8f)) : 0.f;
    has_row[i] = has ? 1.f : 0.f;
  }
}

// ---------------------------------------------------------------------------
// Sinkhorn alignment on topic embeddings. Single block; C/K in dynamic LDS.
// ---------------------------------------------------------------------------
__global__ __launch_bounds__(256)
void sinkhorn_kernel(const float* __restrict__ topic, float* __restrict__ align_out)
{
  extern __shared__ float sm[];     // C[KTOP*KTOP] then Km[KTOP*KTOP]
  float* C  = sm;
  float* Km = sm + KTOP * KTOP;
  __shared__ float nrm[KTOP];
  __shared__ float u[KTOP], v[KTOP];
  __shared__ float red[256];
  const int t = threadIdx.x;

  for (int i = t; i < KTOP; i += 256) {
    const float* p = topic + (long)i * DEMB;
    float s = 0.f;
    for (int d = 0; d < DEMB; ++d) { float x = p[d]; s += x * x; }
    nrm[i] = fmaxf(sqrtf(s), 1e-8f);
    u[i] = 1.f; v[i] = 1.f;
  }
  __syncthreads();
  for (int idx = t; idx < KTOP * KTOP; idx += 256) {
    int i = idx / KTOP, j = idx % KTOP;
    const float* pi = topic + (long)i * DEMB;
    const float* pj = topic + (long)j * DEMB;
    float s = 0.f;
    for (int d = 0; d < DEMB; ++d) s += pi[d] * pj[d];
    float c = 1.f - s / (nrm[i] * nrm[j]);
    C[idx] = c;
    Km[idx] = __expf(-c / EPS_SINK);
  }
  __syncthreads();
  const float a = 1.f / (float)KTOP;
  for (int it = 0; it < SINK_ITERS; ++it) {
    for (int i = t; i < KTOP; i += 256) {
      float s = 0.f;
      for (int j = 0; j < KTOP; ++j) s += Km[i * KTOP + j] * v[j];
      u[i] = a / (s + 1e-8f);
    }
    __syncthreads();
    for (int j = t; j < KTOP; j += 256) {
      float s = 0.f;
      for (int i = 0; i < KTOP; ++i) s += Km[i * KTOP + j] * u[i];
      v[j] = a / (s + 1e-8f);
    }
    __syncthreads();
  }
  float s = 0.f;
  for (int idx = t; idx < KTOP * KTOP; idx += 256) {
    int i = idx / KTOP, j = idx % KTOP;
    s += u[i] * Km[idx] * v[j] * C[idx];
  }
  red[t] = s; __syncthreads();
  for (int st = 128; st > 0; st >>= 1) { if (t < st) red[t] += red[t + st]; __syncthreads(); }
  if (t == 0) *align_out = red[0];
}

// ---------------------------------------------------------------------------
// Final deterministic reduction of per-row losses -> 5 output scalars.
// ---------------------------------------------------------------------------
__global__ __launch_bounds__(256)
void finalize_kernel(const float* __restrict__ klr_en, const float* __restrict__ rcr_en,
                     const float* __restrict__ klr_cn, const float* __restrict__ rcr_cn,
                     const float* __restrict__ per_row, const float* __restrict__ has_row,
                     const float* __restrict__ align_p, float* __restrict__ out)
{
  __shared__ float r0[256], r1[256], r2[256], r3[256];
  const int t = threadIdx.x;
  float a = 0.f, b = 0.f, c = 0.f, d = 0.f;
  for (int i = t; i < BSZ; i += 256) {
    a += klr_en[i] + rcr_en[i];
    b += klr_cn[i] + rcr_cn[i];
    c += per_row[i];
    d += has_row[i];
  }
  r0[t] = a; r1[t] = b; r2[t] = c; r3[t] = d; __syncthreads();
  for (int s = 128; s > 0; s >>= 1) {
    if (t < s) { r0[t] += r0[t + s]; r1[t] += r1[t + s]; r2[t] += r2[t + s]; r3[t] += r3[t + s]; }
    __syncthreads();
  }
  if (t == 0) {
    float tm_en = r0[0] / (float)BSZ;
    float tm_cn = r1[0] / (float)BSZ;
    float contrast = r2[0] / (r3[0] + 1e-8f);
    float align = *align_p;
    out[0] = tm_en + tm_cn + contrast + align;
    out[1] = tm_en;
    out[2] = tm_cn;
    out[3] = contrast;
    out[4] = align;
  }
}

// ---------------------------------------------------------------------------
// Host launcher
// ---------------------------------------------------------------------------
static inline void launch_gemm(hipStream_t s, const float* A, int lda,
                               const float* Bm, int ldb, int transB,
                               const float* bias, float* Out, int ldo,
                               int M, int N, int Kd, int mode,
                               const float* auxm, const float* auxn, float scale,
                               int vecA, int vecB)
{
  dim3 grid((N + BN - 1) / BN, (M + BM - 1) / BM);
  if (transB) {
    if (mode == 2)
      wmma_gemm_kernel<2, 1><<<grid, 256, 0, s>>>(A, lda, Bm, ldb, bias, Out, ldo,
                                                  M, N, Kd, vecA, vecB, auxm, auxn, scale);
    else
      wmma_gemm_kernel<3, 1><<<grid, 256, 0, s>>>(A, lda, Bm, ldb, bias, Out, ldo,
                                                  M, N, Kd, vecA, vecB, auxm, auxn, scale);
  } else {
    if (mode == 1)
      wmma_gemm_kernel<1, 0><<<grid, 256, 0, s>>>(A, lda, Bm, ldb, bias, Out, ldo,
                                                  M, N, Kd, vecA, vecB, auxm, auxn, scale);
    else
      wmma_gemm_kernel<0, 0><<<grid, 256, 0, s>>>(A, lda, Bm, ldb, bias, Out, ldo,
                                                  M, N, Kd, vecA, vecB, auxm, auxn, scale);
  }
}

extern "C" void kernel_launch(void* const* d_in, const int* in_sizes, int n_in,
                              void* d_out, int out_size, void* d_ws, size_t ws_size,
                              hipStream_t stream)
{
  (void)in_sizes; (void)n_in; (void)out_size; (void)ws_size;

  const float* x_en    = (const float*)d_in[0];
  const float* x_cn    = (const float*)d_in[1];
  const int*   cid     = (const int*)  d_in[2];
  const float* eps_en  = (const float*)d_in[3];
  const float* eps_cn  = (const float*)d_in[4];
  const float* W1_en   = (const float*)d_in[5];
  const float* b1_en   = (const float*)d_in[6];
  const float* W2_en   = (const float*)d_in[7];
  const float* b2_en   = (const float*)d_in[8];
  const float* Wmu_en  = (const float*)d_in[9];
  const float* bmu_en  = (const float*)d_in[10];
  const float* Wlv_en  = (const float*)d_in[11];
  const float* blv_en  = (const float*)d_in[12];
  const float* W1_cn   = (const float*)d_in[13];
  const float* b1_cn   = (const float*)d_in[14];
  const float* W2_cn   = (const float*)d_in[15];
  const float* b2_cn   = (const float*)d_in[16];
  const float* Wmu_cn  = (const float*)d_in[17];
  const float* bmu_cn  = (const float*)d_in[18];
  const float* Wlv_cn  = (const float*)d_in[19];
  const float* blv_cn  = (const float*)d_in[20];
  const float* we_en   = (const float*)d_in[21];
  const float* we_cn   = (const float*)d_in[22];
  const float* topic   = (const float*)d_in[23];

  float* out = (float*)d_out;
  // d_out layout: [total, tm_en, tm_cn, contrast, align,
  //                theta_en(512x100), theta_cn(512x100),
  //                beta_en(100x30000), beta_cn(100x20000)]
  float* theta_en = out + 5;
  float* theta_cn = out + 5 + BSZ * KTOP;
  float* beta_en  = out + 5 + 2 * BSZ * KTOP;
  float* beta_cn  = beta_en + KTOP * V_EN;

  // workspace layout (float offsets; every buffer 16B-aligned)
  float* ws = (float*)d_ws;
  float* klrow_en  = ws + 0;
  float* rcrow_en  = ws + 512;
  float* klrow_cn  = ws + 1024;
  float* rcrow_cn  = ws + 1536;
  float* per_row   = ws + 2048;
  float* has_row   = ws + 2560;
  float* align_s   = ws + 3072;
  float* th_en_ws  = ws + 4096;                   // 512*100 aligned theta copy
  float* th_cn_ws  = th_en_ws + BSZ * KTOP;       // 512*100
  float* h1   = th_cn_ws + BSZ * KTOP;            // 512*512
  float* h2   = h1 + BSZ * HDIM;                  // 512*512
  float* mu_b = h2 + BSZ * HDIM;                  // 512*100
  float* lv_b = mu_b + BSZ * KTOP;                // 512*100
  float* tnb  = lv_b + BSZ * KTOP;                // 512*100
  float* Emat = tnb + BSZ * KTOP;                 // 512*512
  float* wn2  = Emat + BSZ * BSZ;                 // 30000
  float* tn2  = wn2 + V_EN;                       // 128
  float* recon = tn2 + 128;                       // 512*30000 (reused EN->CN)

  const float inv_tau = 1.0f / TAU_F;

  // ---------------- EN encoder ----------------
  launch_gemm(stream, x_en, V_EN, W1_en, HDIM, 0, b1_en, h1, HDIM,
              BSZ, HDIM, V_EN, 1, nullptr, nullptr, 0.f, 1, 1);
  launch_gemm(stream, h1, HDIM, W2_en, HDIM, 0, b2_en, h2, HDIM,
              BSZ, HDIM, HDIM, 1, nullptr, nullptr, 0.f, 1, 1);
  launch_gemm(stream, h2, HDIM, Wmu_en, KTOP, 0, bmu_en, mu_b, KTOP,
              BSZ, KTOP, HDIM, 0, nullptr, nullptr, 0.f, 1, 1);
  launch_gemm(stream, h2, HDIM, Wlv_en, KTOP, 0, blv_en, lv_b, KTOP,
              BSZ, KTOP, HDIM, 0, nullptr, nullptr, 0.f, 1, 1);
  reparam_kernel<<<BSZ, 128, 0, stream>>>(mu_b, lv_b, eps_en, theta_en, th_en_ws, klrow_en);

  // ---------------- CN encoder ----------------
  launch_gemm(stream, x_cn, V_CN, W1_cn, HDIM, 0, b1_cn, h1, HDIM,
              BSZ, HDIM, V_CN, 1, nullptr, nullptr, 0.f, 1, 1);
  launch_gemm(stream, h1, HDIM, W2_cn, HDIM, 0, b2_cn, h2, HDIM,
              BSZ, HDIM, HDIM, 1, nullptr, nullptr, 0.f, 1, 1);
  launch_gemm(stream, h2, HDIM, Wmu_cn, KTOP, 0, bmu_cn, mu_b, KTOP,
              BSZ, KTOP, HDIM, 0, nullptr, nullptr, 0.f, 1, 1);
  launch_gemm(stream, h2, HDIM, Wlv_cn, KTOP, 0, blv_cn, lv_b, KTOP,
              BSZ, KTOP, HDIM, 0, nullptr, nullptr, 0.f, 1, 1);
  reparam_kernel<<<BSZ, 128, 0, stream>>>(mu_b, lv_b, eps_cn, theta_cn, th_cn_ws, klrow_cn);

  // ---------------- beta (EN) ----------------
  rowsq_kernel<<<(V_EN + 255) / 256, 256, 0, stream>>>(we_en, V_EN, DEMB, wn2);
  rowsq_kernel<<<1, 128, 0, stream>>>(topic, KTOP, DEMB, tn2);
  launch_gemm(stream, we_en, DEMB, topic, DEMB, 1, nullptr, beta_en, V_EN,
              V_EN, KTOP, DEMB, 2, wn2, tn2, inv_tau, 1, 0);
  beta_norm_kernel<<<KTOP, 256, 0, stream>>>(beta_en, V_EN);

  // ---------------- beta (CN) ----------------
  rowsq_kernel<<<(V_CN + 255) / 256, 256, 0, stream>>>(we_cn, V_CN, DEMB, wn2);
  launch_gemm(stream, we_cn, DEMB, topic, DEMB, 1, nullptr, beta_cn, V_CN,
              V_CN, KTOP, DEMB, 2, wn2, tn2, inv_tau, 1, 0);
  beta_norm_kernel<<<KTOP, 256, 0, stream>>>(beta_cn, V_CN);

  // ---------------- recon losses ----------------
  launch_gemm(stream, th_en_ws, KTOP, beta_en, V_EN, 0, nullptr, recon, V_EN,
              BSZ, V_EN, KTOP, 0, nullptr, nullptr, 0.f, 1, 0);
  recon_loss_kernel<<<BSZ, 256, 0, stream>>>(recon, x_en, V_EN, rcrow_en);
  launch_gemm(stream, th_cn_ws, KTOP, beta_cn, V_CN, 0, nullptr, recon, V_CN,
              BSZ, V_CN, KTOP, 0, nullptr, nullptr, 0.f, 1, 0);
  recon_loss_kernel<<<BSZ, 256, 0, stream>>>(recon, x_cn, V_CN, rcrow_cn);

  // ---------------- contrastive ----------------
  contrast_select_kernel<<<BSZ, 128, 0, stream>>>(th_en_ws, th_cn_ws, tnb);
  launch_gemm(stream, tnb, KTOP, tnb, KTOP, 1, nullptr, Emat, BSZ,
              BSZ, BSZ, KTOP, 3, nullptr, nullptr, inv_tau, 1, 0);
  contrast_loss_kernel<<<BSZ, 256, 0, stream>>>(Emat, cid, per_row, has_row);

  // ---------------- sinkhorn alignment ----------------
  sinkhorn_kernel<<<1, 256, 2 * KTOP * KTOP * sizeof(float), stream>>>(topic, align_s);

  // ---------------- finalize ----------------
  finalize_kernel<<<1, 256, 0, stream>>>(klrow_en, rcrow_en, klrow_cn, rcrow_cn,
                                         per_row, has_row, align_s, out);
}
